// Model_bilstm_34686155882951
// MI455X (gfx1250) — matrix operand order repaired and checked by reading, not compiled, not run on hardware
//
#include <hip/hip_runtime.h>
#include <hip/hip_bf16.h>

// ---------------- problem sizes ----------------
#define Bsz   400
#define Tsz   256
#define EMBD  100
#define Hd    300
#define G4    1200          // 4*H
#define TAGS  18
#define NT    75            // 1200/16 N-tiles
#define MT    25            // 400/16  M-tiles
#define KCX   4             // ceil(100/32) K-chunks for x@Wk (pad to 128)
#define KCH   10            // ceil(300/32) K-chunks for h@Wr (pad to 320)

// ---------------- WMMA types ----------------
typedef __attribute__((ext_vector_type(16))) __bf16 v16bf;
typedef __attribute__((ext_vector_type(8)))  float  v8f;

union Frag16 { v16bf v; uint4 u[2]; };

__device__ __forceinline__ v16bf load_frag16(const __hip_bfloat16* p) {
    Frag16 f;
    const uint4* q = (const uint4*)p;
    f.u[0] = q[0];
    f.u[1] = q[1];
    return f.v;
}

__device__ __forceinline__ float sigmoidf(float x) {
    return 1.0f / (1.0f + __expf(-x));
}

// ======================================================================
// Pack W (Korig x 1200, row-major f32) into B-fragment layout, bf16,
// gate-interleaved columns (N = 4*j + g  <->  orig col = g*300 + j),
// K zero-padded to KC*32.
// Layout: WP[((kc*NT + nt)*32 + lane)*16 + e],
//   K = kc*32 + (lane>>4)*8 + (e>>3)*16 + (e&7),  N = nt*16 + (lane&15)
// ======================================================================
__global__ __launch_bounds__(256)
void pack_w_kernel(const float* __restrict__ W, __hip_bfloat16* __restrict__ WP,
                   int Korig, int KC) {
    int idx = blockIdx.x * blockDim.x + threadIdx.x;
    int total = KC * NT * 32 * 16;
    if (idx >= total) return;
    int e    = idx & 15;
    int lane = (idx >> 4) & 31;
    int nt   = (idx >> 9) % NT;
    int kc   = (idx >> 9) / NT;
    int K    = kc * 32 + (lane >> 4) * 8 + ((e >> 3) << 4) + (e & 7);
    int N    = nt * 16 + (lane & 15);
    int g    = N & 3;
    int j    = N >> 2;
    float v  = (K < Korig) ? W[K * G4 + g * Hd + j] : 0.0f;
    WP[idx]  = __float2bfloat16(v);
}

// gate-interleaved bias: bp[4j+g] = b[g*300+j]
__global__ __launch_bounds__(256)
void perm_bias_kernel(const float* __restrict__ b, float* __restrict__ bp) {
    int N = blockIdx.x * blockDim.x + threadIdx.x;
    if (N < G4) bp[N] = b[(N & 3) * Hd + (N >> 2)];
}

// ======================================================================
// Gather embeddings into A-fragment layout (bf16, K padded 100->128).
// XP[(((t*KCX + kc)*MT + mt)*32 + lane)*16 + e],
//   row b = mt*16 + (lane&15)
//   K = kc*32 + (lane>>4)*8 + (e>>3)*16 + (e&7)
// ======================================================================
__global__ __launch_bounds__(256)
void gather_x_kernel(const int* __restrict__ tokens, const float* __restrict__ emb,
                     __hip_bfloat16* __restrict__ XP) {
    int idx = blockIdx.x * blockDim.x + threadIdx.x;
    int total = Tsz * KCX * MT * 32 * 16;
    if (idx >= total) return;
    int e    = idx & 15;
    int lane = (idx >> 4) & 31;
    int mt   = (idx >> 9) % MT;
    int r2   = (idx >> 9) / MT;
    int kc   = r2 & 3;
    int t    = r2 >> 2;
    int b    = mt * 16 + (lane & 15);
    int K    = kc * 32 + (lane >> 4) * 8 + ((e >> 3) << 4) + (e & 7);
    float v  = 0.0f;
    if (K < EMBD) {
        int tok = tokens[b * Tsz + t];
        v = emb[tok * EMBD + K];
    }
    XP[idx] = __float2bfloat16(v);
}

__global__ __launch_bounds__(256)
void zero_bf16_kernel(__hip_bfloat16* p, int n) {
    int i = blockIdx.x * blockDim.x + threadIdx.x;
    if (i < n) p[i] = __float2bfloat16(0.0f);
}

__global__ __launch_bounds__(256)
void zero_f32_kernel(float* p, int n) {
    int i = blockIdx.x * blockDim.x + threadIdx.x;
    if (i < n) p[i] = 0.0f;
}

// ======================================================================
// One LSTM timestep for one direction.
// Single-wave workgroup computes TWO 16(M) x 16(N) tiles of permuted
//   z = x@Wk + h@Wr + b   via v_wmma_f32_16x16x32_bf16,
// reusing each B fragment (Wk/Wr) across both M tiles (28 WMMA / wave,
// half the B-operand traffic). Since N = 4j+g the tile is gate-complete
// for 4 hidden units: gates + c/h update are done locally via a small
// LDS transpose.  Grid: (NT=75, ceil(MT/2)=13), block: 32 (wave32).
// ======================================================================
__global__ __launch_bounds__(32)
void lstm_step_kernel(const __hip_bfloat16* __restrict__ XP_t,   // A-frags of x at time t
                      const __hip_bfloat16* __restrict__ WkP,    // B-frags, KCX chunks
                      const __hip_bfloat16* __restrict__ WrP,    // B-frags, KCH chunks
                      const float* __restrict__ bperm,
                      const __hip_bfloat16* __restrict__ HprevP, // A-frags of h_{t-1}
                      __hip_bfloat16* __restrict__ HnextP,       // A-frags of h_t (out)
                      float* __restrict__ Cst,                   // [B][300] f32 cell state
                      float* __restrict__ Hout_t,                // Hout + t*B*600 + dir*300
                      const int* __restrict__ tokens, int t) {
    const int  nt   = blockIdx.x;        // 0..74
    const int  mt0  = blockIdx.y * 2;    // 0,2,..,24
    const int  mt1  = mt0 + 1;
    const bool has2 = (mt1 < MT);
    const int  mtb  = has2 ? mt1 : mt0;  // clamp: redundant compute, no OOB
    const int  lane = threadIdx.x;       // 0..31

    v8f acc0 = {};
    v8f acc1 = {};

    // ---- x @ Wk  (K = 0..127, padded) ----
    for (int kc = 0; kc < KCX; ++kc) {
        v16bf bm = load_frag16(WkP  + (((kc * NT + nt)  * 32 + lane) << 4));
        v16bf a0 = load_frag16(XP_t + (((kc * MT + mt0) * 32 + lane) << 4));
        v16bf a1 = load_frag16(XP_t + (((kc * MT + mtb) * 32 + lane) << 4));
        acc0 = __builtin_amdgcn_wmma_f32_16x16x32_bf16(
                   false, a0, false, bm, (short)0, acc0, false, false);
        acc1 = __builtin_amdgcn_wmma_f32_16x16x32_bf16(
                   false, a1, false, bm, (short)0, acc1, false, false);
    }
    // ---- h @ Wr  (K = 0..319, padded) ----
    for (int kc = 0; kc < KCH; ++kc) {
        if (kc + 1 < KCH)   // pull recurrent weights into near caches (WGP scope)
            __builtin_prefetch(WrP + ((((kc + 1) * NT + nt) * 32 + lane) << 4), 0, 3);
        v16bf bm = load_frag16(WrP    + (((kc * NT + nt)  * 32 + lane) << 4));
        v16bf a0 = load_frag16(HprevP + (((kc * MT + mt0) * 32 + lane) << 4));
        v16bf a1 = load_frag16(HprevP + (((kc * MT + mtb) * 32 + lane) << 4));
        acc0 = __builtin_amdgcn_wmma_f32_16x16x32_bf16(
                   false, a0, false, bm, (short)0, acc0, false, false);
        acc1 = __builtin_amdgcn_wmma_f32_16x16x32_bf16(
                   false, a1, false, bm, (short)0, acc1, false, false);
    }

    // ---- spill z tiles to LDS (C layout: vgpr r -> M = (lane>>4)*8 + r, N = lane&15) ----
    __shared__ float zt[2][16][17];
    const int   Nl  = lane & 15;
    const int   hi  = lane >> 4;
    const float bia = bperm[nt * 16 + Nl];
#pragma unroll
    for (int r = 0; r < 8; ++r) {
        zt[0][hi * 8 + r][Nl] = acc0[r] + bia;
        zt[1][hi * 8 + r][Nl] = acc1[r] + bia;
    }
    __syncthreads();   // single-wave workgroup: lowers to S_NOP, acts as fence

    // ---- gates + state update: per tile 64 (m,j) cells, 2 per lane ----
    for (int u = 0; u < 2; ++u) {
        if (u == 1 && !has2) break;
        const int mt = u ? mt1 : mt0;
#pragma unroll
        for (int p = 0; p < 2; ++p) {
            int cell = lane * 2 + p;    // 0..63
            int ml   = cell >> 2;       // local row 0..15
            int jl   = cell & 3;        // local hidden unit 0..3
            float zi = zt[u][ml][4 * jl + 0];
            float zf = zt[u][ml][4 * jl + 1];
            float zg = zt[u][ml][4 * jl + 2];
            float zo = zt[u][ml][4 * jl + 3];

            int b = mt * 16 + ml;
            int j = nt * 4 + jl;        // global hidden index 0..299

            float ig = sigmoidf(zi);
            float fg = sigmoidf(zf);
            float og = sigmoidf(zo);
            float gg = tanhf(zg);

            float cprev = Cst[b * Hd + j];
            float cnew  = fg * cprev + ig * gg;
            float hnew  = og * tanhf(cnew);

            // locate (b, j) inside the packed A-fragment h buffer
            int kc  = j >> 5;
            int r   = j & 31;
            int hh  = (r >> 3) & 1;
            int e   = (r & 7) + ((r >> 4) << 3);
            int pix = (((kc * MT + mt) * 32 + (hh * 16 + ml)) << 4) + e;

            float hprev = __bfloat162float(HprevP[pix]);
            bool  msk   = tokens[b * Tsz + t] != 0;
            float h = msk ? hnew : hprev;
            float c = msk ? cnew : cprev;

            Cst[b * Hd + j]      = c;
            Hout_t[b * 600 + j]  = h;
            HnextP[pix]          = __float2bfloat16(h);
        }
    }
}

// ======================================================================
// Final dense: out[b][t][k] = concat(h_fw,h_bw)[t][b][:] @ W(600x18) + b
// ======================================================================
__global__ __launch_bounds__(256)
void dense_kernel(const float* __restrict__ Hout, const float* __restrict__ W,
                  const float* __restrict__ bias, float* __restrict__ out) {
    __shared__ float ws[600 * TAGS];
    for (int i = threadIdx.x; i < 600 * TAGS; i += 256) ws[i] = W[i];
    __syncthreads();

    int idx = blockIdx.x * 256 + threadIdx.x;   // b*T + t
    if (idx >= Bsz * Tsz) return;
    int bb = idx / Tsz;
    int tt = idx % Tsz;
    const float* hrow = Hout + ((size_t)tt * Bsz + bb) * 600;

    float accv[TAGS];
#pragma unroll
    for (int k = 0; k < TAGS; ++k) accv[k] = bias[k];
    for (int j = 0; j < 600; ++j) {
        float hv = hrow[j];
        const float* wr = &ws[j * TAGS];
#pragma unroll
        for (int k = 0; k < TAGS; ++k) accv[k] += hv * wr[k];
    }
    float* orow = out + (size_t)idx * TAGS;
#pragma unroll
    for (int k = 0; k < TAGS; ++k) orow[k] = accv[k];
}

// ======================================================================
extern "C" void kernel_launch(void* const* d_in, const int* in_sizes, int n_in,
                              void* d_out, int out_size, void* d_ws, size_t ws_size,
                              hipStream_t stream) {
    (void)in_sizes; (void)n_in; (void)out_size; (void)ws_size;

    const int*   tokens  = (const int*)  d_in[0];
    const float* emb     = (const float*)d_in[1];
    const float* fw_k    = (const float*)d_in[2];
    const float* fw_r    = (const float*)d_in[3];
    const float* fw_b    = (const float*)d_in[4];
    const float* bw_k    = (const float*)d_in[5];
    const float* bw_r    = (const float*)d_in[6];
    const float* bw_b    = (const float*)d_in[7];
    const float* dense_w = (const float*)d_in[8];
    const float* dense_b = (const float*)d_in[9];

    // ---- carve workspace ----
    char*  base = (char*)d_ws;
    size_t off  = 0;
    auto carve = [&](size_t bytes) -> char* {
        off = (off + 255) & ~(size_t)255;
        char* p = base + off;
        off += bytes;
        return p;
    };
    const size_t XP_ELEMS  = (size_t)Tsz * KCX * MT * 32 * 16;   // 13,107,200
    const size_t WKP_ELEMS = (size_t)KCX * NT * 32 * 16;         // 153,600
    const size_t WRP_ELEMS = (size_t)KCH * NT * 32 * 16;         // 384,000
    const size_t HB_ELEMS  = (size_t)KCH * MT * 32 * 16;         // 128,000 per buffer

    __hip_bfloat16* XP    = (__hip_bfloat16*)carve(XP_ELEMS * 2);
    __hip_bfloat16* WkPf  = (__hip_bfloat16*)carve(WKP_ELEMS * 2);
    __hip_bfloat16* WkPb  = (__hip_bfloat16*)carve(WKP_ELEMS * 2);
    __hip_bfloat16* WrPf  = (__hip_bfloat16*)carve(WRP_ELEMS * 2);
    __hip_bfloat16* WrPb  = (__hip_bfloat16*)carve(WRP_ELEMS * 2);
    float*          bpf   = (float*)carve(G4 * 4);
    float*          bpb   = (float*)carve(G4 * 4);
    __hip_bfloat16* Hball = (__hip_bfloat16*)carve(4 * HB_ELEMS * 2); // fw0,fw1,bw0,bw1
    float*          Call  = (float*)carve(2 * (size_t)Bsz * Hd * 4);  // fw, bw
    float*          Hout  = (float*)carve((size_t)Tsz * Bsz * 600 * 4);

    __hip_bfloat16* Hbf[2] = { Hball,                Hball + HB_ELEMS     };
    __hip_bfloat16* Hbb[2] = { Hball + 2 * HB_ELEMS, Hball + 3 * HB_ELEMS };
    float* Cf = Call;
    float* Cb = Call + (size_t)Bsz * Hd;

    // ---- prep: pack weights / bias, zero state ----
    pack_w_kernel<<<(int)((WKP_ELEMS + 255) / 256), 256, 0, stream>>>(fw_k, WkPf, EMBD, KCX);
    pack_w_kernel<<<(int)((WKP_ELEMS + 255) / 256), 256, 0, stream>>>(bw_k, WkPb, EMBD, KCX);
    pack_w_kernel<<<(int)((WRP_ELEMS + 255) / 256), 256, 0, stream>>>(fw_r, WrPf, Hd,   KCH);
    pack_w_kernel<<<(int)((WRP_ELEMS + 255) / 256), 256, 0, stream>>>(bw_r, WrPb, Hd,   KCH);
    perm_bias_kernel<<<(G4 + 255) / 256, 256, 0, stream>>>(fw_b, bpf);
    perm_bias_kernel<<<(G4 + 255) / 256, 256, 0, stream>>>(bw_b, bpb);
    zero_bf16_kernel<<<(int)((4 * HB_ELEMS + 255) / 256), 256, 0, stream>>>(Hball, (int)(4 * HB_ELEMS));
    zero_f32_kernel<<<(2 * Bsz * Hd + 255) / 256, 256, 0, stream>>>(Call, 2 * Bsz * Hd);

    // ---- gather embeddings into A-fragment layout ----
    gather_x_kernel<<<(int)((XP_ELEMS + 255) / 256), 256, 0, stream>>>(tokens, emb, XP);

    // ---- sequential bidirectional scan ----
    const size_t XSTEP = (size_t)KCX * MT * 32 * 16;   // frag elems per timestep
    dim3 sgrid(NT, (MT + 1) / 2);                      // (75, 13)
    for (int s = 0; s < Tsz; ++s) {
        int tf = s;
        int tb = Tsz - 1 - s;
        lstm_step_kernel<<<sgrid, 32, 0, stream>>>(
            XP + (size_t)tf * XSTEP, WkPf, WrPf, bpf,
            Hbf[s & 1], Hbf[(s + 1) & 1], Cf,
            Hout + (size_t)tf * Bsz * 600 + 0, tokens, tf);
        lstm_step_kernel<<<sgrid, 32, 0, stream>>>(
            XP + (size_t)tb * XSTEP, WkPb, WrPb, bpb,
            Hbb[s & 1], Hbb[(s + 1) & 1], Cb,
            Hout + (size_t)tb * Bsz * 600 + Hd, tokens, tb);
    }

    // ---- output projection ----
    dense_kernel<<<(Bsz * Tsz + 255) / 256, 256, 0, stream>>>(Hout, dense_w, dense_b,
                                                              (float*)d_out);
}
